// Pointnet2Seg_68753836474379
// MI455X (gfx1250) — matrix operand order repaired and checked by reading, not compiled, not run on hardware
//
#include <hip/hip_runtime.h>
#include <hip/hip_bf16.h>

// ---------------------------------------------------------------------------
// PointNet++ segmentation forward, CDNA5 (gfx1250, wave32).
// Pointwise convs -> bf16 WMMA GEMMs (v_wmma_f32_16x16x32_bf16, f32 accum),
// double-buffered LDS staging with b128 global loads / b64 LDS stores.
// Train-mode BN: GEMM fuses bias + per-channel sum/sumsq; tiny finalize kernel;
// fused scale/shift+ReLU float4 pass.
// ---------------------------------------------------------------------------

typedef __bf16 bf16_t;
typedef __attribute__((ext_vector_type(16))) __bf16 bf16x16;
typedef __attribute__((ext_vector_type(8)))  float  f32x8;

#define GTM 128            // block M tile
#define GTN 32             // block N tile
#define GTK 32             // K step
#define LDSK 36            // padded bf16 row stride (72 B, 8B-aligned rows)

// ---------------------------------------------------------------------------
// GEMM: Y[M,Cout] = X[M,Cin] * W[Cout,Cin]^T + bias, fused per-channel
// sum/sumsq into stats[2*Cout]. Requires M%128==0, Cin%32==0, Cout%32==0.
// 8 waves: waveM = wave&3 (32 rows each), waveN = wave>>2 (16 cols each);
// each wave computes 32x16 via two WMMAs sharing one B fragment.
// ---------------------------------------------------------------------------
__global__ __launch_bounds__(256)
void gemm_bn_stats(const float* __restrict__ X, const float* __restrict__ W,
                   const float* __restrict__ bias, float* __restrict__ Y,
                   float* __restrict__ stats, int M, int Cin, int Cout)
{
    __shared__ __align__(16) bf16_t sX[2][GTM][LDSK];
    __shared__ __align__(16) bf16_t sW[2][GTN][LDSK];
    __shared__ float ssum[GTN];
    __shared__ float ssq[GTN];

    const int tid   = threadIdx.x;
    const int lane  = tid & 31;
    const int wave  = tid >> 5;
    const int waveM = wave & 3;
    const int waveN = wave >> 2;
    const int m0    = blockIdx.x * GTM;
    const int n0    = blockIdx.y * GTN;
    const int mrow  = lane & 15;
    const int half  = lane >> 4;

    if (tid < GTN) { ssum[tid] = 0.f; ssq[tid] = 0.f; }

    const float* Xb = X + m0 * Cin;   // max flat index ~33.5M -> int math safe

    // Per-thread staged tiles: X 128x32 floats = 1024 float4 (4/thread),
    // W 32x32 floats = 256 float4 (1/thread).
    float4 rx[4];
    float4 rw;

    auto fetch = [&](int k0) {
        #pragma unroll
        for (int j = 0; j < 4; ++j) {
            int f = tid + 256 * j;
            int r = f >> 3, c4 = (f & 7) << 2;
            rx[j] = *(const float4*)&Xb[r * Cin + k0 + c4];
        }
        {
            int r = tid >> 3, c4 = (tid & 7) << 2;
            rw = *(const float4*)&W[(n0 + r) * Cin + k0 + c4];
        }
    };
    auto stash = [&](int p) {
        #pragma unroll
        for (int j = 0; j < 4; ++j) {
            int f = tid + 256 * j;
            int r = f >> 3, c4 = (f & 7) << 2;
            union { bf16_t h[4]; unsigned long long u64; } cv;
            cv.h[0] = (bf16_t)rx[j].x; cv.h[1] = (bf16_t)rx[j].y;
            cv.h[2] = (bf16_t)rx[j].z; cv.h[3] = (bf16_t)rx[j].w;
            *(unsigned long long*)&sX[p][r][c4] = cv.u64;
        }
        {
            int r = tid >> 3, c4 = (tid & 7) << 2;
            union { bf16_t h[4]; unsigned long long u64; } cv;
            cv.h[0] = (bf16_t)rw.x; cv.h[1] = (bf16_t)rw.y;
            cv.h[2] = (bf16_t)rw.z; cv.h[3] = (bf16_t)rw.w;
            *(unsigned long long*)&sW[p][r][c4] = cv.u64;
        }
    };

    f32x8 acc0 = {0.f,0.f,0.f,0.f,0.f,0.f,0.f,0.f};
    f32x8 acc1 = {0.f,0.f,0.f,0.f,0.f,0.f,0.f,0.f};

    fetch(0);
    stash(0);
    __syncthreads();

    int p = 0;
    for (int k0 = 0; k0 < Cin; k0 += GTK) {
        const int kn = k0 + GTK;
        if (kn < Cin) fetch(kn);          // overlap next-tile VMEM with WMMA

        // Fragments per ISA 7.12.2 (A: lane=M, K pairs split 0..7/8..15 by
        // lane half; B: lane=N, same K split as A).
        const bf16_t* arow0 = &sX[p][waveM * 32 + mrow][0];
        const bf16_t* arow1 = &sX[p][waveM * 32 + 16 + mrow][0];
        const bf16_t* brow  = &sW[p][waveN * 16 + mrow][0];
        bf16x16 a0, a1, b;
        #pragma unroll
        for (int v = 0; v < 8; ++v) {
            int ka = ((v < 4) ? 0 : 16) + half * 8 + (v & 3) * 2;
            a0[2 * v]     = arow0[ka];
            a0[2 * v + 1] = arow0[ka + 1];
            a1[2 * v]     = arow1[ka];
            a1[2 * v + 1] = arow1[ka + 1];
            int kb = ((v < 4) ? 0 : 16) + half * 8 + (v & 3) * 2;
            b[2 * v]     = brow[kb];
            b[2 * v + 1] = brow[kb + 1];
        }
        acc0 = __builtin_amdgcn_wmma_f32_16x16x32_bf16(false, a0, false, b,
                                                       (short)0, acc0, false, false);
        acc1 = __builtin_amdgcn_wmma_f32_16x16x32_bf16(false, a1, false, b,
                                                       (short)0, acc1, false, false);

        if (kn < Cin) stash(p ^ 1);
        __syncthreads();
        p ^= 1;
    }

    // C/D layout: VGPR r -> M = r (lanes 0-15) / r+8 (lanes 16-31); N = lane&15.
    const int   n  = waveN * 16 + (lane & 15);
    const float bv = bias[n0 + n];
    float s1 = 0.f, s2 = 0.f;
    #pragma unroll
    for (int r = 0; r < 8; ++r) {
        int   mA = m0 + waveM * 32 + half * 8 + r;
        float y0 = acc0[r] + bv;
        float y1 = acc1[r] + bv;
        Y[(size_t)mA * Cout + n0 + n]        = y0;
        Y[(size_t)(mA + 16) * Cout + n0 + n] = y1;
        s1 += y0 + y1;
        s2 += y0 * y0 + y1 * y1;
    }
    atomicAdd(&ssum[n], s1);
    atomicAdd(&ssq[n],  s2);
    __syncthreads();
    if (tid < GTN) {
        atomicAdd(&stats[n0 + tid],        ssum[tid]);
        atomicAdd(&stats[Cout + n0 + tid], ssq[tid]);
    }
}

// ---------------------------------------------------------------------------
// Small helpers
// ---------------------------------------------------------------------------
__global__ void zero_buf(float* __restrict__ p, int n)
{
    int i = blockIdx.x * blockDim.x + threadIdx.x;
    if (i < n) p[i] = 0.f;
}

__global__ void bn_finalize(float* __restrict__ stats, const float* __restrict__ g,
                            const float* __restrict__ be, int C, float Mf)
{
    int c = threadIdx.x;
    if (c >= C) return;
    float mean  = stats[c] / Mf;
    float var   = stats[C + c] / Mf - mean * mean;
    float scale = g[c] * rsqrtf(var + 1e-5f);
    stats[c]     = scale;
    stats[C + c] = be[c] - mean * scale;
}

// float4 fused BN scale/shift + ReLU, in place. total4 = M*C/4, C%4==0.
__global__ void bn_apply_relu(float* __restrict__ Y, const float* __restrict__ stats,
                              size_t total4, int C)
{
    size_t i = (size_t)blockIdx.x * blockDim.x + threadIdx.x;
    if (i >= total4) return;
    size_t e = i * 4;
    int    c = (int)(e % C);
    float4 y  = *(float4*)&Y[e];
    float4 sc = *(const float4*)&stats[c];
    float4 sh = *(const float4*)&stats[C + c];
    y.x = fmaxf(y.x * sc.x + sh.x, 0.f);
    y.y = fmaxf(y.y * sc.y + sh.y, 0.f);
    y.z = fmaxf(y.z * sc.z + sh.z, 0.f);
    y.w = fmaxf(y.w * sc.w + sh.w, 0.f);
    *(float4*)&Y[e] = y;
}

// Tiny-Cin pointwise conv (emb1: 3->64) with fused stats. blockDim.x == Cout.
__global__ void pw_small_stats(const float* __restrict__ X, const float* __restrict__ W,
                               const float* __restrict__ bias, float* __restrict__ Y,
                               float* __restrict__ stats, int M, int Cin, int Cout)
{
    int c  = threadIdx.x;
    int r0 = blockIdx.x * 32;
    float wv[8];
    for (int k = 0; k < Cin; ++k) wv[k] = W[(size_t)c * Cin + k];
    float bv = bias[c];
    float s1 = 0.f, s2 = 0.f;
    for (int rr = 0; rr < 32; ++rr) {
        int r = r0 + rr;
        if (r >= M) break;
        float acc = bv;
        for (int k = 0; k < Cin; ++k) acc += X[(size_t)r * Cin + k] * wv[k];
        Y[(size_t)r * Cout + c] = acc;
        s1 += acc;
        s2 += acc * acc;
    }
    atomicAdd(&stats[c], s1);
    atomicAdd(&stats[Cout + c], s2);
}

// Final linear head (128 -> 8), no BN/ReLU.
__global__ void pw_plain(const float* __restrict__ X, const float* __restrict__ W,
                         const float* __restrict__ bias, float* __restrict__ Y,
                         int M, int Cin, int Cout)
{
    size_t i = (size_t)blockIdx.x * blockDim.x + threadIdx.x;
    if (i >= (size_t)M * Cout) return;
    int c = (int)(i % Cout);
    int r = (int)(i / Cout);
    float acc = bias[c];
    for (int k = 0; k < Cin; ++k) acc += X[(size_t)r * Cin + k] * W[(size_t)c * Cin + k];
    Y[i] = acc;
}

// ---------------------------------------------------------------------------
// Farthest point sampling: one block per batch, dist[] resident in LDS.
// ---------------------------------------------------------------------------
__global__ __launch_bounds__(1024)
void fps_kernel(const float* __restrict__ xyz, int* __restrict__ out_idx, int N, int S)
{
    __shared__ float dist[8192];
    __shared__ float rmax[1024];
    __shared__ int   rarg[1024];
    __shared__ int   cur;

    int b = blockIdx.x, t = threadIdx.x, T = blockDim.x;
    const float* P = xyz + (size_t)b * N * 3;

    for (int i = t; i < N; i += T) dist[i] = 1e10f;
    if (t == 0) cur = 0;
    __syncthreads();

    for (int s = 0; s < S; ++s) {
        int c = cur;
        if (t == 0) out_idx[(size_t)b * S + s] = c;
        float cx = P[c * 3], cy = P[c * 3 + 1], cz = P[c * 3 + 2];
        float best = -1.f;
        int   bi   = 0;
        for (int i = t; i < N; i += T) {
            float dx = P[i * 3] - cx, dy = P[i * 3 + 1] - cy, dz = P[i * 3 + 2] - cz;
            float d  = dx * dx + dy * dy + dz * dz;
            float dd = fminf(dist[i], d);
            dist[i]  = dd;
            if (dd > best) { best = dd; bi = i; }
        }
        rmax[t] = best; rarg[t] = bi;
        __syncthreads();
        for (int o = T >> 1; o > 0; o >>= 1) {
            if (t < o && rmax[t + o] > rmax[t]) { rmax[t] = rmax[t + o]; rarg[t] = rarg[t + o]; }
            __syncthreads();
        }
        if (t == 0) cur = rarg[0];
        __syncthreads();
    }
}

__global__ void gather_xyz(const float* __restrict__ xyz, const int* __restrict__ idx,
                           float* __restrict__ out, int N, int S, size_t total)
{
    size_t i = (size_t)blockIdx.x * blockDim.x + threadIdx.x;
    if (i >= total) return;
    int c = (int)(i % 3);
    int s = (int)((i / 3) % S);
    int b = (int)(i / ((size_t)3 * S));
    out[i] = xyz[((size_t)b * N + idx[(size_t)b * S + s]) * 3 + c];
}

// ---------------------------------------------------------------------------
// K nearest neighbors by iterative argmin; candidate distances cached in LDS.
// One block (256 thr) per query; used for K=32 grouping and K=3 interpolation.
// ---------------------------------------------------------------------------
__global__ __launch_bounds__(256)
void knn_kernel(const float* __restrict__ Q, const float* __restrict__ Cnd,
                int* __restrict__ oidx, float* __restrict__ odist,
                int Nq, int Nc, int K)
{
    __shared__ float d[8192];
    __shared__ float rmin[256];
    __shared__ int   rarg[256];

    int t = threadIdx.x;
    int q = blockIdx.x % Nq;
    int b = blockIdx.x / Nq;

    const float* qp = Q + ((size_t)b * Nq + q) * 3;
    float qx = qp[0], qy = qp[1], qz = qp[2];
    const float* cp = Cnd + (size_t)b * Nc * 3;

    for (int i = t; i < Nc; i += 256) {
        float dx = cp[i * 3] - qx, dy = cp[i * 3 + 1] - qy, dz = cp[i * 3 + 2] - qz;
        d[i] = dx * dx + dy * dy + dz * dz;
    }
    __syncthreads();

    for (int k = 0; k < K; ++k) {
        float best = 3.4e38f;
        int   bi   = 0;
        for (int i = t; i < Nc; i += 256) {
            float v = d[i];
            if (v < best) { best = v; bi = i; }
        }
        rmin[t] = best; rarg[t] = bi;
        __syncthreads();
        for (int o = 128; o > 0; o >>= 1) {
            if (t < o && rmin[t + o] < rmin[t]) { rmin[t] = rmin[t + o]; rarg[t] = rarg[t + o]; }
            __syncthreads();
        }
        if (t == 0) {
            int sel = rarg[0];
            oidx[((size_t)b * Nq + q) * K + k] = sel;
            if (odist) odist[((size_t)b * Nq + q) * K + k] = rmin[0];
            d[sel] = 3.4e38f;
        }
        __syncthreads();
    }
}

// G[b,s,k,:] = concat(feat[knn[k]] - feat[fps[s]], feat[fps[s]]), float4 over c.
__global__ void group_concat(const float* __restrict__ feat, const int* __restrict__ fidx,
                             const int* __restrict__ kidx, float* __restrict__ G,
                             int Nn, int S, int K, int C, size_t total4)
{
    size_t i = (size_t)blockIdx.x * blockDim.x + threadIdx.x;
    if (i >= total4) return;
    size_t e    = i * 4;
    int    C2   = 2 * C;
    int    c2   = (int)(e % C2);
    size_t rest = e / C2;
    int    k    = (int)(rest % K); rest /= K;
    int    s    = (int)(rest % S);
    int    b    = (int)(rest / S);
    int    ctr  = fidx[(size_t)b * S + s];
    const float* fb = feat + (size_t)b * Nn * C;
    int    cc   = (c2 < C) ? c2 : c2 - C;
    float4 cv   = *(const float4*)&fb[(size_t)ctr * C + cc];
    if (c2 < C) {
        int    nb = kidx[((size_t)b * S + s) * K + k];
        float4 nv = *(const float4*)&fb[(size_t)nb * C + c2];
        cv.x = nv.x - cv.x; cv.y = nv.y - cv.y; cv.z = nv.z - cv.z; cv.w = nv.w - cv.w;
    }
    *(float4*)&G[e] = cv;
}

// out[b,s,c4] = max_k in[b,s,k,c4], float4 over channels.
__global__ void max_over_k(const float* __restrict__ in, float* __restrict__ out,
                           size_t total4 /*B*S*C/4*/, int K, int C)
{
    size_t i = (size_t)blockIdx.x * blockDim.x + threadIdx.x;
    if (i >= total4) return;
    int    C4   = C >> 2;
    int    c    = (int)(i % C4) << 2;
    size_t bs   = i / C4;
    size_t base = bs * (size_t)K * C + c;
    float4 m = *(const float4*)&in[base];
    for (int k = 1; k < K; ++k) {
        float4 v = *(const float4*)&in[base + (size_t)k * C];
        m.x = fmaxf(m.x, v.x); m.y = fmaxf(m.y, v.y);
        m.z = fmaxf(m.z, v.z); m.w = fmaxf(m.w, v.w);
    }
    *(float4*)&out[bs * C + c] = m;
}

// Out[b,n,:] = concat(P1[b,n,:], sum_j w_j * P2[b,idx3[j],:]), float4 over c.
__global__ void interp_concat(const float* __restrict__ P1, const float* __restrict__ P2,
                              const int* __restrict__ idx3, const float* __restrict__ dst3,
                              float* __restrict__ Out, int N, int S, int C1, int C2,
                              size_t total4)
{
    size_t i = (size_t)blockIdx.x * blockDim.x + threadIdx.x;
    if (i >= total4) return;
    size_t e  = i * 4;
    int    Ct = C1 + C2;
    int    c  = (int)(e % Ct);
    int    n  = (int)((e / Ct) % N);
    int    b  = (int)(e / ((size_t)Ct * N));
    float4 r;
    if (c < C1) {
        r = *(const float4*)&P1[((size_t)b * N + n) * C1 + c];
    } else {
        size_t base = ((size_t)b * N + n) * 3;
        float  w0 = 1.f / (dst3[base] + 1e-8f);
        float  w1 = 1.f / (dst3[base + 1] + 1e-8f);
        float  w2 = 1.f / (dst3[base + 2] + 1e-8f);
        float  wi = 1.f / (w0 + w1 + w2);
        int    c2 = c - C1;
        const float* p2b = P2 + (size_t)b * S * C2;
        float4 v0 = *(const float4*)&p2b[(size_t)idx3[base]     * C2 + c2];
        float4 v1 = *(const float4*)&p2b[(size_t)idx3[base + 1] * C2 + c2];
        float4 v2 = *(const float4*)&p2b[(size_t)idx3[base + 2] * C2 + c2];
        r.x = (w0 * v0.x + w1 * v1.x + w2 * v2.x) * wi;
        r.y = (w0 * v0.y + w1 * v1.y + w2 * v2.y) * wi;
        r.z = (w0 * v0.z + w1 * v1.z + w2 * v2.z) * wi;
        r.w = (w0 * v0.w + w1 * v1.w + w2 * v2.w) * wi;
    }
    *(float4*)&Out[e] = r;
}

// ---------------------------------------------------------------------------
// Host-side orchestration
// ---------------------------------------------------------------------------
struct Lp { const float *w, *b, *g, *be; };

extern "C" void kernel_launch(void* const* d_in, const int* in_sizes, int n_in,
                              void* d_out, int out_size, void* d_ws, size_t ws_size,
                              hipStream_t stream)
{
    (void)in_sizes; (void)n_in; (void)out_size; (void)ws_size;

    const int Bn = 4, N = 8192, S1 = 2048, S2 = 1024, KN = 32;
    const int M0 = Bn * N;      // 32768
    const int M1 = Bn * S1;     // 8192
    const float* x = (const float*)d_in[0];
    int p = 1;
    auto takeL = [&](bool bn) {
        Lp L;
        L.w = (const float*)d_in[p++];
        L.b = (const float*)d_in[p++];
        if (bn) { L.g = (const float*)d_in[p++]; L.be = (const float*)d_in[p++]; }
        else    { L.g = nullptr; L.be = nullptr; }
        return L;
    };
    Lp emb1 = takeL(true), emb2 = takeL(true);
    Lp loc01 = takeL(true), loc02 = takeL(true);
    Lp loc11 = takeL(true), loc12 = takeL(true);
    Lp fp01 = takeL(true), fp02 = takeL(true);
    Lp fp11 = takeL(true), fp12 = takeL(true);
    Lp head1 = takeL(true), head2 = takeL(true);
    Lp outL = takeL(false);

    float* wsf = (float*)d_ws;
    size_t off = 0;
    auto allocf = [&](size_t n) { float* q = wsf + off; off += n; return q; };

    float* bufA  = allocf((size_t)Bn * S1 * KN * 128);
    float* bufB  = allocf((size_t)Bn * S1 * KN * 128);
    float* feat0 = allocf((size_t)Bn * N * 64);
    float* feat1 = allocf((size_t)Bn * S1 * 128);
    float* feat2 = allocf((size_t)Bn * S2 * 256);
    float* hfp0  = allocf((size_t)Bn * S1 * 512);
    float* cat0  = allocf((size_t)Bn * S1 * 384);
    float* cat1  = allocf((size_t)Bn * N * 576);
    float* xyz1  = allocf((size_t)Bn * S1 * 3);
    float* xyz2  = allocf((size_t)Bn * S2 * 3);
    float* dst30 = allocf((size_t)Bn * S1 * 3);
    float* dst31 = allocf((size_t)Bn * N * 3);
    float* stats = allocf(2048);
    int* fps1  = (int*)allocf((size_t)Bn * S1);
    int* fps2  = (int*)allocf((size_t)Bn * S2);
    int* knn1  = (int*)allocf((size_t)Bn * S1 * KN);
    int* knn2  = (int*)allocf((size_t)Bn * S2 * KN);
    int* idx30 = (int*)allocf((size_t)Bn * S1 * 3);
    int* idx31 = (int*)allocf((size_t)Bn * N * 3);

    auto cbr = [&](const float* Xf, const Lp& L, float* Yf, int M, int Cin, int Cout) {
        zero_buf<<<(2 * Cout + 255) / 256, 256, 0, stream>>>(stats, 2 * Cout);
        dim3 grid(M / GTM, Cout / GTN);
        gemm_bn_stats<<<grid, 256, 0, stream>>>(Xf, L.w, L.b, Yf, stats, M, Cin, Cout);
        bn_finalize<<<1, Cout, 0, stream>>>(stats, L.g, L.be, Cout, (float)M);
        size_t t4 = (size_t)M * Cout / 4;
        bn_apply_relu<<<(unsigned)((t4 + 255) / 256), 256, 0, stream>>>(Yf, stats, t4, Cout);
    };

    // ---- embedding: 3 -> 64 -> 64 ------------------------------------------
    zero_buf<<<1, 128, 0, stream>>>(stats, 128);
    pw_small_stats<<<M0 / 32, 64, 0, stream>>>(x, emb1.w, emb1.b, bufA, stats, M0, 3, 64);
    bn_finalize<<<1, 64, 0, stream>>>(stats, emb1.g, emb1.be, 64, (float)M0);
    {
        size_t t4 = (size_t)M0 * 64 / 4;
        bn_apply_relu<<<(unsigned)((t4 + 255) / 256), 256, 0, stream>>>(bufA, stats, t4, 64);
    }
    cbr(bufA, emb2, feat0, M0, 64, 64);

    // ---- stage 1: sample+group (2048,32) + local op 128->128->128 ----------
    fps_kernel<<<Bn, 1024, 0, stream>>>(x, fps1, N, S1);
    {
        size_t tot = (size_t)Bn * S1 * 3;
        gather_xyz<<<(unsigned)((tot + 255) / 256), 256, 0, stream>>>(x, fps1, xyz1, N, S1, tot);
    }
    knn_kernel<<<Bn * S1, 256, 0, stream>>>(xyz1, x, knn1, nullptr, S1, N, KN);
    {
        size_t t4 = (size_t)Bn * S1 * KN * 128 / 4;
        group_concat<<<(unsigned)((t4 + 255) / 256), 256, 0, stream>>>(
            feat0, fps1, knn1, bufA, N, S1, KN, 64, t4);
    }
    cbr(bufA, loc01, bufB, Bn * S1 * KN, 128, 128);
    cbr(bufB, loc02, bufA, Bn * S1 * KN, 128, 128);
    {
        size_t t4 = (size_t)Bn * S1 * 128 / 4;
        max_over_k<<<(unsigned)((t4 + 255) / 256), 256, 0, stream>>>(bufA, feat1, t4, KN, 128);
    }

    // ---- stage 2: sample+group (1024,32) + local op 256->256->256 ----------
    fps_kernel<<<Bn, 1024, 0, stream>>>(xyz1, fps2, S1, S2);
    {
        size_t tot = (size_t)Bn * S2 * 3;
        gather_xyz<<<(unsigned)((tot + 255) / 256), 256, 0, stream>>>(xyz1, fps2, xyz2, S1, S2, tot);
    }
    knn_kernel<<<Bn * S2, 256, 0, stream>>>(xyz2, xyz1, knn2, nullptr, S2, S1, KN);
    {
        size_t t4 = (size_t)Bn * S2 * KN * 256 / 4;
        group_concat<<<(unsigned)((t4 + 255) / 256), 256, 0, stream>>>(
            feat1, fps2, knn2, bufA, S1, S2, KN, 128, t4);
    }
    cbr(bufA, loc11, bufB, Bn * S2 * KN, 256, 256);
    cbr(bufB, loc12, bufA, Bn * S2 * KN, 256, 256);
    {
        size_t t4 = (size_t)Bn * S2 * 256 / 4;
        max_over_k<<<(unsigned)((t4 + 255) / 256), 256, 0, stream>>>(bufA, feat2, t4, KN, 256);
    }

    // ---- feature prop 0: (xyz1 <- xyz2), 384 -> 256 -> 512 -----------------
    knn_kernel<<<Bn * S1, 256, 0, stream>>>(xyz1, xyz2, idx30, dst30, S1, S2, 3);
    {
        size_t t4 = (size_t)Bn * S1 * 384 / 4;
        interp_concat<<<(unsigned)((t4 + 255) / 256), 256, 0, stream>>>(
            feat1, feat2, idx30, dst30, cat0, S1, S2, 128, 256, t4);
    }
    cbr(cat0, fp01, bufB, M1, 384, 256);
    cbr(bufB, fp02, hfp0, M1, 256, 512);

    // ---- feature prop 1: (xyz <- xyz1), 576 -> 512 -> 1024 -----------------
    knn_kernel<<<Bn * N, 256, 0, stream>>>(x, xyz1, idx31, dst31, N, S1, 3);
    {
        size_t t4 = (size_t)Bn * N * 576 / 4;
        interp_concat<<<(unsigned)((t4 + 255) / 256), 256, 0, stream>>>(
            feat0, hfp0, idx31, dst31, cat1, N, S1, 64, 512, t4);
    }
    cbr(cat1, fp11, bufA, M0, 576, 512);
    cbr(bufA, fp12, bufB, M0, 512, 1024);

    // ---- head: 1024 -> 512 -> 128 -> 8 -------------------------------------
    cbr(bufB, head1, bufA, M0, 1024, 512);
    cbr(bufA, head2, bufB, M0, 512, 128);
    {
        size_t tot = (size_t)M0 * 8;
        pw_plain<<<(unsigned)((tot + 255) / 256), 256, 0, stream>>>(
            bufB, outL.w, outL.b, (float*)d_out, M0, 128, 8);
    }
}
